// LlamaAttention_83854941487680
// MI455X (gfx1250) — matrix operand order repaired and checked by reading, not compiled, not run on hardware
//
#include <hip/hip_runtime.h>
#include <hip/hip_bf16.h>
#include <cstdint>
#include <cstddef>

// ---------------------------------------------------------------- types
typedef __attribute__((ext_vector_type(16))) __bf16       v16bf;
typedef __attribute__((ext_vector_type(8)))  float        v8f;
typedef __attribute__((ext_vector_type(4)))  unsigned int v4ui;

union FragU { v16bf bf; v4ui u[2]; };

// ---------------------------------------------------------------- problem dims
constexpr int BB   = 2;
constexpr int SS   = 2048;
constexpr int HIDN = 2048;
constexpr int NH   = 16;
constexpr int HD   = 128;   // head dim after concat
constexpr int HD2  = 64;    // nope/rope half
constexpr int LAT  = 256;
constexpr int ROWS = BB * SS;   // 4096

// ---------------------------------------------------------------- CDNA5 async copy
// GLOBAL_LOAD_ASYNC_TO_LDS_B128: per-lane global -> LDS, tracked by ASYNCcnt.
__device__ __forceinline__ void async_copy_b128(const __bf16* gsrc,
                                                __bf16* lds_dst) {
  unsigned lds = (unsigned)(uintptr_t)lds_dst;
  asm volatile("global_load_async_to_lds_b128 %0, %1, off"
               :: "v"(lds), "v"(gsrc)
               : "memory");
}
__device__ __forceinline__ void wait_async() {
  asm volatile("s_wait_asynccnt 0x0" ::: "memory");
}

// ---------------------------------------------------------------- WMMA helpers
__device__ __forceinline__ v8f wmma_bf16(v16bf a, v16bf b, v8f c) {
  return __builtin_amdgcn_wmma_f32_16x16x32_bf16(
      false, a, false, b, (short)0, c, false, false);
}

// A-matrix 16x32 bf16 fragment, row-major source, leading dim ld (elems).
// Lane layout: m = lane%16, g = lane/16; regs 0..7 hold K = g*16..g*16+15
// contiguous  ->  two contiguous 16B loads.
__device__ __forceinline__ v16bf load_A_vec(const __bf16* base, int ld) {
  int l = threadIdx.x & 31;
  int m = l & 15, g = l >> 4;
  const __bf16* p = base + (size_t)m * ld + g * 16;
  FragU f;
  f.u[0] = *(const v4ui*)(p);
  f.u[1] = *(const v4ui*)(p + 8);
  return f.bf;
}

// B fragment from pre-swizzled storage: packed[lane][16] contiguous bf16.
__device__ __forceinline__ v16bf load_B_packed(const __bf16* base) {
  int l = threadIdx.x & 31;
  const __bf16* p = base + l * 16;
  FragU f;
  f.u[0] = *(const v4ui*)(p);
  f.u[1] = *(const v4ui*)(p + 8);
  return f.bf;
}

// B^T fragment: logical B[k][n] = src[n*ld + k] (used for Q@K^T with K stored
// [key][d]).  Per lane: n = lane%16, K run g*16..g*16+15 contiguous along d.
__device__ __forceinline__ v16bf load_BT_vec(const __bf16* base, int ld) {
  int l = threadIdx.x & 31;
  int n = l & 15, g = l >> 4;
  const __bf16* p = base + (size_t)n * ld + g * 16;
  FragU f;
  f.u[0] = *(const v4ui*)(p);
  f.u[1] = *(const v4ui*)(p + 8);
  return f.bf;
}

// ---------------------------------------------------------------- convert
__global__ void k_f32_to_bf16(const float* __restrict__ in,
                              __bf16* __restrict__ out, size_t n) {
  size_t i = (size_t)blockIdx.x * blockDim.x + threadIdx.x;
  if (i < n) out[i] = (__bf16)in[i];
}

// ---------------------------------------------------------------- generic GEMM
// C[M,N] = A[M,K] @ W[K,N], bf16 in, f32 accumulate.
// Block tile 128x128, 256 thr = 8 waves in 4x2; wave tile 32x64 (2m x 4n frags).
// OUT_F32 selects the epilogue store type at compile time (no branches).
constexpr int GBM = 128, GBN = 128, GBK = 32;

template <bool OUT_F32>
__global__ void __launch_bounds__(256)
k_gemm_bf16(const __bf16* __restrict__ A, const __bf16* __restrict__ W,
            void* __restrict__ outp, int M, int N, int K) {
  __shared__ __align__(16) __bf16 As[GBM][GBK];          // 8 KB, row-major
  __shared__ __align__(16) __bf16 Bsw[GBN / 16][32][16]; // 8 KB, fragment order
  const int bm = blockIdx.x * GBM;
  const int bn = blockIdx.y * GBN;
  const int wave = threadIdx.x >> 5;
  const int wr = wave >> 1;   // 0..3  (row group of 32)
  const int wc = wave & 1;    // 0..1  (col group of 64)

  v8f acc[2][4];
#pragma unroll
  for (int mi = 0; mi < 2; ++mi)
#pragma unroll
    for (int nt = 0; nt < 4; ++nt) acc[mi][nt] = v8f{};

  // B swizzle thread mapping (exactly 256 pair-chunks per tile)
  const int kp_b = threadIdx.x >> 4;         // 0..15 -> k = 2*kp
  const int n0_b = (threadIdx.x & 15) * 8;   // column octet

  for (int k0 = 0; k0 < K; k0 += GBK) {
    // A tile 128x32: async global->LDS b128, no VGPR round trip
#pragma unroll
    for (int i = threadIdx.x; i < GBM * GBK / 8; i += 256) {
      int r = i >> 2, c = (i & 3) * 8;
      async_copy_b128(&A[(size_t)(bm + r) * K + k0 + c], &As[r][c]);
    }
    // B tile: two coalesced b128 row-loads, scatter 8 packed u32 to LDS
    {
      int k = kp_b * 2;
      v4ui r0 = *(const v4ui*)&W[(size_t)(k0 + k) * N + bn + n0_b];
      v4ui r1 = *(const v4ui*)&W[(size_t)(k0 + k + 1) * N + bn + n0_b];
      const unsigned short* a0 = (const unsigned short*)&r0;
      const unsigned short* a1 = (const unsigned short*)&r1;
#pragma unroll
      for (int e = 0; e < 8; ++e) {
        int n = n0_b + e;
        *(unsigned int*)&Bsw[n >> 4][((k >> 4) << 4) | (n & 15)][k & 15] =
            (unsigned int)a0[e] | ((unsigned int)a1[e] << 16);
      }
    }
    wait_async();
    __syncthreads();

    v16bf af0 = load_A_vec(&As[wr * 32][0], GBK);
    v16bf af1 = load_A_vec(&As[wr * 32 + 16][0], GBK);
#pragma unroll
    for (int nt = 0; nt < 4; ++nt) {
      v16bf bfrag = load_B_packed(&Bsw[wc * 4 + nt][0][0]);
      acc[0][nt] = wmma_bf16(af0, bfrag, acc[0][nt]);
      acc[1][nt] = wmma_bf16(af1, bfrag, acc[1][nt]);
    }
    __syncthreads();
  }

  const int l = threadIdx.x & 31;
  const int col = l & 15, half = l >> 4;
  float*  outF = (float*)outp;
  __bf16* outB = (__bf16*)outp;
#pragma unroll
  for (int mi = 0; mi < 2; ++mi) {
#pragma unroll
    for (int nt = 0; nt < 4; ++nt) {
#pragma unroll
      for (int r = 0; r < 8; ++r) {
        size_t m = (size_t)(unsigned)(bm + wr * 32 + mi * 16 + r + 8 * half);
        size_t n = (size_t)(unsigned)(bn + wc * 64 + nt * 16 + col);
        if constexpr (OUT_F32) outF[m * N + n] = acc[mi][nt][r];
        else                   outB[m * N + n] = (__bf16)acc[mi][nt][r];
      }
    }
  }
}

// ---------------------------------------------------------------- RoPE + concat
__global__ void k_rope_concat(const __bf16* __restrict__ p_part,
                              const __bf16* __restrict__ r_part,
                              __bf16* __restrict__ out) {
  size_t idx = (size_t)blockIdx.x * blockDim.x + threadIdx.x;
  size_t total = (size_t)ROWS * NH * HD2;
  if (idx >= total) return;
  int d = (int)(idx & 63);
  size_t t = idx >> 6;
  int h = (int)(t & (NH - 1));
  size_t row = t >> 4;
  int s = (int)(row & (SS - 1));

  size_t src = row * (NH * HD2) + h * HD2;
  size_t dst = (row * NH + h) * HD;

  out[dst + d] = p_part[src + d];

  int j = (d < 32) ? d : d - 32;
  float inv_freq = __expf(-((float)(2 * j) / 64.0f) * 9.210340371976184f);
  float ang = (float)s * inv_freq;
  float sn = __sinf(ang), cs = __cosf(ang);
  float x1 = (float)r_part[src + j];
  float x2 = (float)r_part[src + j + 32];
  float val = (d < 32) ? (x1 * cs - x2 * sn) : (x1 * sn + x2 * cs);
  out[dst + HD2 + d] = (__bf16)val;
}

// ---------------------------------------------------------------- flash attention
// Q,K,V,O: [B, S, H, 128] bf16.  Block = (b,h, 64-row q tile), 128 thr = 4 waves.
constexpr int FTM = 64, FTN = 64;

__global__ void __launch_bounds__(128)
k_flash_attn(const __bf16* __restrict__ Q, const __bf16* __restrict__ Kg,
             const __bf16* __restrict__ Vg, __bf16* __restrict__ O) {
  __shared__ __align__(16) __bf16 Ks[FTN][HD];             // 16 KB row-major
  __shared__ __align__(16) __bf16 Vsw[HD / 16][2][32][16]; // 16 KB fragment order
  __shared__ __align__(16) float  Sld[FTM][FTN];           // 16 KB
  __shared__ __align__(16) __bf16 Pld[FTM][FTN];           // 8 KB
  __shared__ float mrow[FTM], lrow[FTM], arow[FTM];

  const int bh = blockIdx.x;
  const int b = bh / NH, h = bh % NH;
  const int qt = blockIdx.y;
  const int qbase = qt * FTM;
  const int wave = threadIdx.x >> 5;
  const int l = threadIdx.x & 31;
  const int col = l & 15, half = l >> 4;
  const float scale = 0.08838834764831845f;   // 1/sqrt(128)

  // Q fragments in registers (4 K-slices of wave's 16x128 strip)
  const __bf16* Qrow = Q + ((size_t)(b * SS + qbase + wave * 16) * NH + h) * HD;
  v16bf qf[4];
#pragma unroll
  for (int kk = 0; kk < 4; ++kk) qf[kk] = load_A_vec(Qrow + kk * 32, NH * HD);

  if (threadIdx.x < FTM) { mrow[threadIdx.x] = -1e30f; lrow[threadIdx.x] = 0.f; }

  v8f acc[8];
#pragma unroll
  for (int nt = 0; nt < 8; ++nt) acc[nt] = v8f{};

  const size_t bs_base = (size_t)b * SS;

  const int nkt = qt + 1;
  for (int kt = 0; kt < nkt; ++kt) {
    const int kbase = kt * FTN;
    __syncthreads();
    // K tile row-major: async global->LDS b128 (8 issues per thread)
    const size_t gk = ((bs_base + kbase) * NH + h) * (size_t)HD;
#pragma unroll
    for (int i = threadIdx.x; i < FTN * HD / 8; i += 128) {
      int r = i >> 4, c = (i & 15) * 8;
      async_copy_b128(&Kg[gk + (size_t)r * (NH * HD) + c], &Ks[r][c]);
    }
    // V tile swizzled: coalesced b128 loads of key pairs, packed u32 scatter
#pragma unroll
    for (int pc = threadIdx.x; pc < (FTN / 2) * (HD / 8); pc += 128) {
      int kp = pc >> 4;           // key pair 0..31
      int d0 = (pc & 15) * 8;     // d octet
      int k  = kp * 2;
      size_t g0 = ((bs_base + kbase + k) * NH + h) * (size_t)HD + d0;
      v4ui r0 = *(const v4ui*)&Vg[g0];
      v4ui r1 = *(const v4ui*)&Vg[g0 + (size_t)NH * HD];
      const unsigned short* a0 = (const unsigned short*)&r0;
      const unsigned short* a1 = (const unsigned short*)&r1;
#pragma unroll
      for (int e = 0; e < 8; ++e) {
        int d = d0 + e;
        *(unsigned int*)&Vsw[d >> 4][k >> 5][(((k >> 4) & 1) << 4) | (d & 15)]
                            [k & 15] = (unsigned int)a0[e] |
                                       ((unsigned int)a1[e] << 16);
      }
    }
    wait_async();
    __syncthreads();

    // S strip = Q @ K^T
#pragma unroll
    for (int nt = 0; nt < 4; ++nt) {
      v8f c = v8f{};
#pragma unroll
      for (int kk = 0; kk < 4; ++kk) {
        v16bf bfrag = load_BT_vec(&Ks[nt * 16][kk * 32], HD);
        c = wmma_bf16(qf[kk], bfrag, c);
      }
#pragma unroll
      for (int r = 0; r < 8; ++r) {
        int mloc = wave * 16 + r + 8 * half;
        int mglb = qbase + mloc;
        int nglb = kbase + nt * 16 + col;
        float sv = c[r] * scale;
        if (nglb > mglb) sv = -1e9f;
        Sld[mloc][nt * 16 + col] = sv;
      }
    }
    __syncthreads();

    // online softmax, one thread per row, float4 reads, u32-packed P writes
    if (threadIdx.x < FTM) {
      int rr = threadIdx.x;
      const float4* rowp = (const float4*)&Sld[rr][0];
      float mo = mrow[rr];
      float mx = mo;
#pragma unroll
      for (int j4 = 0; j4 < FTN / 4; ++j4) {
        float4 v = rowp[j4];
        mx = fmaxf(mx, fmaxf(fmaxf(v.x, v.y), fmaxf(v.z, v.w)));
      }
      float al = __expf(mo - mx);
      float ssum = 0.f;
      unsigned int* prow = (unsigned int*)&Pld[rr][0];
#pragma unroll
      for (int j2 = 0; j2 < FTN / 2; ++j2) {
        float p0 = __expf(Sld[rr][2 * j2] - mx);
        float p1 = __expf(Sld[rr][2 * j2 + 1] - mx);
        __bf16 b0 = (__bf16)p0, b1 = (__bf16)p1;
        prow[j2] = (unsigned int)*(unsigned short*)&b0 |
                   ((unsigned int)*(unsigned short*)&b1 << 16);
        ssum += p0 + p1;
      }
      mrow[rr] = mx;
      lrow[rr] = lrow[rr] * al + ssum;
      arow[rr] = al;
    }
    __syncthreads();

    // rescale accumulators, then acc += P @ V
#pragma unroll
    for (int r = 0; r < 8; ++r) {
      float al = arow[wave * 16 + r + 8 * half];
#pragma unroll
      for (int nt = 0; nt < 8; ++nt) acc[nt][r] *= al;
    }
#pragma unroll
    for (int kk = 0; kk < 2; ++kk) {
      v16bf afrag = load_A_vec(&Pld[wave * 16][kk * 32], FTN);
#pragma unroll
      for (int nt = 0; nt < 8; ++nt) {
        v16bf bfrag = load_B_packed(&Vsw[nt][kk][0][0]);
        acc[nt] = wmma_bf16(afrag, bfrag, acc[nt]);
      }
    }
  }
  __syncthreads();

  // normalize + store ctx
#pragma unroll
  for (int r = 0; r < 8; ++r) {
    int mloc = wave * 16 + r + 8 * half;
    float inv = 1.0f / lrow[mloc];
    size_t orow = ((bs_base + qbase + mloc) * NH + h) * (size_t)HD;
#pragma unroll
    for (int nt = 0; nt < 8; ++nt)
      O[orow + nt * 16 + col] = (__bf16)(acc[nt][r] * inv);
  }
}

// ---------------------------------------------------------------- host
static inline void* wsptr(void* base, size_t& off, size_t bytes) {
  void* p = (char*)base + off;
  off += (bytes + 255) & ~(size_t)255;
  return p;
}

extern "C" void kernel_launch(void* const* d_in, const int* in_sizes, int n_in,
                              void* d_out, int out_size, void* d_ws, size_t ws_size,
                              hipStream_t stream) {
  (void)in_sizes; (void)n_in; (void)out_size; (void)ws_size;

  const float* hid  = (const float*)d_in[0];
  const float* wkvd = (const float*)d_in[1];
  const float* wqd  = (const float*)d_in[2];
  const float* wku  = (const float*)d_in[3];
  const float* wqu  = (const float*)d_in[4];
  const float* wvu  = (const float*)d_in[5];
  const float* wrk  = (const float*)d_in[6];
  const float* wrq  = (const float*)d_in[7];
  const float* wo   = (const float*)d_in[8];
  float* out = (float*)d_out;

  size_t off = 0;
  __bf16* hid_bf  = (__bf16*)wsptr(d_ws, off, (size_t)ROWS * HIDN * 2);
  __bf16* wkvd_bf = (__bf16*)wsptr(d_ws, off, (size_t)HIDN * LAT * 2);
  __bf16* wqd_bf  = (__bf16*)wsptr(d_ws, off, (size_t)HIDN * LAT * 2);
  __bf16* wku_bf  = (__bf16*)wsptr(d_ws, off, (size_t)LAT * NH * HD2 * 2);
  __bf16* wqu_bf  = (__bf16*)wsptr(d_ws, off, (size_t)LAT * NH * HD2 * 2);
  __bf16* wvu_bf  = (__bf16*)wsptr(d_ws, off, (size_t)LAT * NH * HD * 2);
  __bf16* wrk_bf  = (__bf16*)wsptr(d_ws, off, (size_t)HIDN * NH * HD2 * 2);
  __bf16* wrq_bf  = (__bf16*)wsptr(d_ws, off, (size_t)LAT * NH * HD2 * 2);
  __bf16* wo_bf   = (__bf16*)wsptr(d_ws, off, (size_t)HIDN * HIDN * 2);
  __bf16* kvd_bf  = (__bf16*)wsptr(d_ws, off, (size_t)ROWS * LAT * 2);
  __bf16* qd_bf   = (__bf16*)wsptr(d_ws, off, (size_t)ROWS * LAT * 2);
  __bf16* kp_bf   = (__bf16*)wsptr(d_ws, off, (size_t)ROWS * NH * HD2 * 2);
  __bf16* qp_bf   = (__bf16*)wsptr(d_ws, off, (size_t)ROWS * NH * HD2 * 2);
  __bf16* kr_bf   = (__bf16*)wsptr(d_ws, off, (size_t)ROWS * NH * HD2 * 2);
  __bf16* qr_bf   = (__bf16*)wsptr(d_ws, off, (size_t)ROWS * NH * HD2 * 2);
  __bf16* v_bf    = (__bf16*)wsptr(d_ws, off, (size_t)ROWS * NH * HD * 2);
  __bf16* k_bf    = (__bf16*)wsptr(d_ws, off, (size_t)ROWS * NH * HD * 2);
  __bf16* q_bf    = (__bf16*)wsptr(d_ws, off, (size_t)ROWS * NH * HD * 2);
  __bf16* ctx_bf  = (__bf16*)wsptr(d_ws, off, (size_t)ROWS * NH * HD * 2);

  auto cvt = [&](const float* src, __bf16* dst, size_t n) {
    k_f32_to_bf16<<<(unsigned)((n + 255) / 256), 256, 0, stream>>>(src, dst, n);
  };
  cvt(hid,  hid_bf,  (size_t)ROWS * HIDN);
  cvt(wkvd, wkvd_bf, (size_t)HIDN * LAT);
  cvt(wqd,  wqd_bf,  (size_t)HIDN * LAT);
  cvt(wku,  wku_bf,  (size_t)LAT * NH * HD2);
  cvt(wqu,  wqu_bf,  (size_t)LAT * NH * HD2);
  cvt(wvu,  wvu_bf,  (size_t)LAT * NH * HD);
  cvt(wrk,  wrk_bf,  (size_t)HIDN * NH * HD2);
  cvt(wrq,  wrq_bf,  (size_t)LAT * NH * HD2);
  cvt(wo,   wo_bf,   (size_t)HIDN * HIDN);

  auto gemmB = [&](const __bf16* A, const __bf16* W, __bf16* o,
                   int M, int N, int K) {
    dim3 grid(M / GBM, N / GBN);
    k_gemm_bf16<false><<<grid, 256, 0, stream>>>(A, W, o, M, N, K);
  };
  auto gemmF = [&](const __bf16* A, const __bf16* W, float* o,
                   int M, int N, int K) {
    dim3 grid(M / GBM, N / GBN);
    k_gemm_bf16<true><<<grid, 256, 0, stream>>>(A, W, o, M, N, K);
  };

  gemmB(hid_bf, wkvd_bf, kvd_bf, ROWS, LAT, HIDN);
  gemmB(hid_bf, wqd_bf,  qd_bf,  ROWS, LAT, HIDN);
  gemmB(kvd_bf, wku_bf,  kp_bf,  ROWS, NH * HD2, LAT);
  gemmB(qd_bf,  wqu_bf,  qp_bf,  ROWS, NH * HD2, LAT);
  gemmB(kvd_bf, wvu_bf,  v_bf,   ROWS, NH * HD,  LAT);
  gemmB(hid_bf, wrk_bf,  kr_bf,  ROWS, NH * HD2, HIDN);
  gemmB(qd_bf,  wrq_bf,  qr_bf,  ROWS, NH * HD2, LAT);

  {
    size_t total = (size_t)ROWS * NH * HD2;
    unsigned g = (unsigned)((total + 255) / 256);
    k_rope_concat<<<g, 256, 0, stream>>>(kp_bf, kr_bf, k_bf);
    k_rope_concat<<<g, 256, 0, stream>>>(qp_bf, qr_bf, q_bf);
  }

  {
    dim3 grid(BB * NH, SS / FTM);
    k_flash_attn<<<grid, 128, 0, stream>>>(q_bf, k_bf, v_bf, ctx_bf);
  }

  gemmF(ctx_bf, wo_bf, out, ROWS, HIDN, HIDN);
}
